// ProtDiGCNEncoderDecoder_7335804142314
// MI455X (gfx1250) — matrix operand order repaired and checked by reading, not compiled, not run on hardware
//
#include <hip/hip_runtime.h>
#include <math.h>

#define N_NODES 10000
#define N_EDGES 640000
#define F_IN 256
#define H1 256
#define H2 128

typedef __attribute__((ext_vector_type(16))) __bf16 v16bf;
typedef __attribute__((ext_vector_type(8)))  __bf16 v8bf;
typedef __attribute__((ext_vector_type(8)))  float  v8f;

__device__ __forceinline__ unsigned short f32_to_bf16_bits(float f) {
  unsigned int u = __float_as_uint(f);
  u += 0x7fffu + ((u >> 16) & 1u);   // round-to-nearest-even
  return (unsigned short)(u >> 16);
}

// ---------------------------------------------------------------------------
// Positional encoding add + f32 -> bf16 convert.  idx over N_NODES*F_IN exact.
// ---------------------------------------------------------------------------
__global__ void pe_add_bf16_kernel(const float* __restrict__ x,
                                   unsigned short* __restrict__ xb) {
  int idx = blockIdx.x * blockDim.x + threadIdx.x;
  int f   = idx & (F_IN - 1);
  int pos = f >> 6;          // n-gram position 0..3
  int d   = f & 63;          // pe dim
  // div = exp((d & ~1) * (-ln(10000)/64))
  float dv  = __expf((float)(d & ~1) * (-9.210340371976184f / 64.0f));
  float ang = (float)pos * dv;
  float pe  = (d & 1) ? __cosf(ang) : __sinf(ang);
  xb[idx] = f32_to_bf16_bits(x[idx] + pe);
}

// ---------------------------------------------------------------------------
// out_bf16 = bf16(a + b)   (b == nullptr -> plain convert)
// ---------------------------------------------------------------------------
__global__ void addcvt_bf16_kernel(const float* __restrict__ a,
                                   const float* __restrict__ b,
                                   unsigned short* __restrict__ o, int n) {
  int i = blockIdx.x * blockDim.x + threadIdx.x;
  if (i >= n) return;
  float v = a[i];
  if (b) v += b[i];
  o[i] = f32_to_bf16_bits(v);
}

// ---------------------------------------------------------------------------
// D[M,N] = A[M,K](bf16) @ B[N,K]^T(bf16) + bias[N]   via v_wmma_f32_16x16x32_bf16
// One 16x16 tile per wave32; M,N multiples of 16; K multiple of 32.
// A-operand: lane m=lane&15; lanes 0-15 K 0-15, lanes 16-31 K 16-31 (contiguous, same order as B)
// B-operand: lane n=lane&15; lanes 0-15 K 0-15, lanes 16-31 K 16-31 (contiguous in [N,K])
// ---------------------------------------------------------------------------
__global__ void gemm_bf16_wmma_kernel(const unsigned short* __restrict__ A,
                                      const unsigned short* __restrict__ B,
                                      const float* __restrict__ bias,
                                      float* __restrict__ D,
                                      int M, int N, int K) {
  const int lane   = threadIdx.x & 31;
  const int tilesN = N >> 4;
  const int tiles  = (M >> 4) * tilesN;
  const int tile   = blockIdx.x * (blockDim.x >> 5) + (threadIdx.x >> 5);
  if (tile >= tiles) return;                      // wave-uniform: EXEC stays all-1s
  const int tm = tile / tilesN;
  const int tn = tile - tm * tilesN;

  const int half = lane >> 4;                     // 0: lanes 0-15, 1: lanes 16-31
  const int l15  = lane & 15;
  const size_t arow = (size_t)(tm * 16 + l15) * K + (half ? 16 : 0);
  const size_t brow = (size_t)(tn * 16 + l15) * K + (half ? 16 : 0);

  v8f acc = {};
  for (int k0 = 0; k0 < K; k0 += 32) {
    // A: two 16-byte slabs (K = kb..kb+7 and kb+8..kb+15)
    v8bf a_lo = *reinterpret_cast<const v8bf*>(A + arow + k0);
    v8bf a_hi = *reinterpret_cast<const v8bf*>(A + arow + k0 + 8);
    v16bf av  = __builtin_shufflevector(a_lo, a_hi,
                  0,1,2,3,4,5,6,7,8,9,10,11,12,13,14,15);
    // B: one contiguous 32-byte slab (K = kb..kb+15)
    v16bf bv = *reinterpret_cast<const v16bf*>(B + brow + k0);

    if (k0 + 32 < K) {                            // global_prefetch_b8 next slab
      __builtin_prefetch(A + arow + k0 + 32, 0, 1);
      __builtin_prefetch(B + brow + k0 + 32, 0, 1);
    }
    acc = __builtin_amdgcn_wmma_f32_16x16x32_bf16(
        /*neg_a=*/false, av, /*neg_b=*/false, bv,
        /*c_mod=*/(short)0, acc, /*reuse_a=*/false, /*reuse_b=*/false);
  }

  const int col   = tn * 16 + l15;
  const float bvs = bias ? bias[col] : 0.0f;
  const int rbase = tm * 16 + half * 8;           // lanes 16-31 hold rows M=8..15
#pragma unroll
  for (int r = 0; r < 8; ++r)
    D[(size_t)(rbase + r) * N + col] = acc[r] + bvs;
}

// ---------------------------------------------------------------------------
// Scatter-add edge propagation: y[dst,:] += w_e * t[src,:]   (f32 atomics)
// One block per edge, threads over features (coalesced rows).
// ---------------------------------------------------------------------------
__global__ void edge_prop_kernel(const float* __restrict__ t,
                                 const int* __restrict__ eidx,
                                 const float* __restrict__ w,
                                 float* __restrict__ y,
                                 int nEdges, int F) {
  const int e   = blockIdx.x;
  const int src = eidx[e];
  const int dst = eidx[nEdges + e];
  const float we = w[e];
  const float* ts = t + (size_t)src * F;
  float*       yd = y + (size_t)dst * F;
  for (int f = threadIdx.x; f < F; f += blockDim.x)
    atomicAdd(&yd[f], we * ts[f]);
}

// ---------------------------------------------------------------------------
// h = tanh(Cin*(yin+bmi+bsi) + Cout*(yout+bmo+bso)); optional pre-tanh embed out.
// ---------------------------------------------------------------------------
__global__ void layer_combine_kernel(const float* __restrict__ yin,
                                     const float* __restrict__ yout,
                                     const float* __restrict__ bmi,
                                     const float* __restrict__ bsi,
                                     const float* __restrict__ bmo,
                                     const float* __restrict__ bso,
                                     const float* __restrict__ Cin,
                                     const float* __restrict__ Cout,
                                     unsigned short* __restrict__ h_bf16,
                                     float* __restrict__ embed,
                                     int F) {
  int idx = blockIdx.x * blockDim.x + threadIdx.x;
  int n = idx / F;
  int f = idx - n * F;
  float v = Cin[n]  * (yin[idx]  + bmi[f] + bsi[f])
          + Cout[n] * (yout[idx] + bmo[f] + bso[f]);
  if (embed) embed[idx] = v;
  h_bf16[idx] = f32_to_bf16_bits(tanhf(v));
}

// ---------------------------------------------------------------------------
// In-place row-wise log_softmax; one 256-thread block (8 wave32) per row.
// ---------------------------------------------------------------------------
__global__ void log_softmax_kernel(float* __restrict__ logits, int C) {
  float* p = logits + (size_t)blockIdx.x * C;
  const int tid = threadIdx.x;
  __shared__ float red[8];

  float m = -3.402823466e38f;
  for (int i = tid; i < C; i += blockDim.x) m = fmaxf(m, p[i]);
  for (int o = 16; o > 0; o >>= 1) m = fmaxf(m, __shfl_down(m, o, 32));
  if ((tid & 31) == 0) red[tid >> 5] = m;
  __syncthreads();
  if (tid == 0) {
    float t = red[0];
    for (int i = 1; i < 8; ++i) t = fmaxf(t, red[i]);
    red[0] = t;
  }
  __syncthreads();
  m = red[0];
  __syncthreads();

  float s = 0.0f;
  for (int i = tid; i < C; i += blockDim.x) s += __expf(p[i] - m);
  for (int o = 16; o > 0; o >>= 1) s += __shfl_down(s, o, 32);
  if ((tid & 31) == 0) red[tid >> 5] = s;
  __syncthreads();
  if (tid == 0) {
    float t = 0.0f;
    for (int i = 0; i < 8; ++i) t += red[i];
    red[0] = t;
  }
  __syncthreads();
  const float lse = m + __logf(red[0]);
  for (int i = tid; i < C; i += blockDim.x) p[i] -= lse;
}

// ---------------------------------------------------------------------------
extern "C" void kernel_launch(void* const* d_in, const int* in_sizes, int n_in,
                              void* d_out, int out_size, void* d_ws, size_t ws_size,
                              hipStream_t stream) {
  (void)in_sizes; (void)n_in; (void)out_size; (void)ws_size;

  const float* x    = (const float*)d_in[0];
  const int*   ein  = (const int*)  d_in[1];
  const float* win  = (const float*)d_in[2];
  const int*   eout = (const int*)  d_in[3];
  const float* wout = (const float*)d_in[4];
  const float* W1mi = (const float*)d_in[5];
  const float* W1mo = (const float*)d_in[6];
  const float* W1s  = (const float*)d_in[7];
  const float* b1mi = (const float*)d_in[8];
  const float* b1mo = (const float*)d_in[9];
  const float* b1si = (const float*)d_in[10];
  const float* b1so = (const float*)d_in[11];
  const float* C1i  = (const float*)d_in[12];
  const float* C1o  = (const float*)d_in[13];
  const float* W2mi = (const float*)d_in[14];
  const float* W2mo = (const float*)d_in[15];
  const float* W2s  = (const float*)d_in[16];
  const float* b2mi = (const float*)d_in[17];
  const float* b2mo = (const float*)d_in[18];
  const float* b2si = (const float*)d_in[19];
  const float* b2so = (const float*)d_in[20];
  const float* C2i  = (const float*)d_in[21];
  const float* C2o  = (const float*)d_in[22];
  const float* Wd   = (const float*)d_in[23];
  const float* bd   = (const float*)d_in[24];

  char* ws = (char*)d_ws;
  size_t off = 0;
  auto alloc = [&](size_t bytes) {
    size_t o = off; off += (bytes + 255) & ~(size_t)255; return o;
  };
  // persistent scratch
  unsigned short* xb   = (unsigned short*)(ws + alloc((size_t)N_NODES * F_IN * 2));
  unsigned short* w1ci = (unsigned short*)(ws + alloc((size_t)H1 * F_IN * 2));
  unsigned short* w1co = (unsigned short*)(ws + alloc((size_t)H1 * F_IN * 2));
  unsigned short* w2ci = (unsigned short*)(ws + alloc((size_t)H2 * H1 * 2));
  unsigned short* w2co = (unsigned short*)(ws + alloc((size_t)H2 * H1 * 2));
  unsigned short* wdb  = (unsigned short*)(ws + alloc((size_t)N_NODES * H2 * 2));
  unsigned short* h1b  = (unsigned short*)(ws + alloc((size_t)N_NODES * H1 * 2));
  unsigned short* h2b  = (unsigned short*)(ws + alloc((size_t)N_NODES * H2 * 2));
  // layer-1 scratch; layer-2 aliases it (dead by then)
  size_t scratchBase = off;
  float* t1i = (float*)(ws + alloc((size_t)N_NODES * H1 * 4));
  float* t1o = (float*)(ws + alloc((size_t)N_NODES * H1 * 4));
  float* y1i = (float*)(ws + alloc((size_t)N_NODES * H1 * 4));
  float* y1o = (float*)(ws + alloc((size_t)N_NODES * H1 * 4));
  off = scratchBase;
  float* t2i = (float*)(ws + alloc((size_t)N_NODES * H2 * 4));
  float* t2o = (float*)(ws + alloc((size_t)N_NODES * H2 * 4));
  float* y2i = (float*)(ws + alloc((size_t)N_NODES * H2 * 4));
  float* y2o = (float*)(ws + alloc((size_t)N_NODES * H2 * 4));

  auto gemm = [&](const unsigned short* A, const unsigned short* B,
                  const float* bias, float* D, int M, int N, int K) {
    int tiles  = (M / 16) * (N / 16);
    int blocks = (tiles + 7) / 8;                 // 8 wave32 per 256-thread block
    gemm_bf16_wmma_kernel<<<blocks, 256, 0, stream>>>(A, B, bias, D, M, N, K);
  };

  // --- input: positional encoding + bf16 convert; weight combine (Wm + Wskip)
  pe_add_bf16_kernel<<<N_NODES * F_IN / 256, 256, 0, stream>>>(x, xb);
  int nw1 = H1 * F_IN, nw2 = H2 * H1, nwd = N_NODES * H2;
  addcvt_bf16_kernel<<<(nw1 + 255) / 256, 256, 0, stream>>>(W1mi, W1s, w1ci, nw1);
  addcvt_bf16_kernel<<<(nw1 + 255) / 256, 256, 0, stream>>>(W1mo, W1s, w1co, nw1);
  addcvt_bf16_kernel<<<(nw2 + 255) / 256, 256, 0, stream>>>(W2mi, W2s, w2ci, nw2);
  addcvt_bf16_kernel<<<(nw2 + 255) / 256, 256, 0, stream>>>(W2mo, W2s, w2co, nw2);
  addcvt_bf16_kernel<<<(nwd + 255) / 256, 256, 0, stream>>>(Wd, nullptr, wdb, nwd);

  // --- layer 1: GEMM (combined weights) -> edge scatter -> combine/tanh
  gemm(xb, w1ci, nullptr, t1i, N_NODES, H1, F_IN);
  gemm(xb, w1co, nullptr, t1o, N_NODES, H1, F_IN);
  hipMemsetAsync(y1i, 0, (size_t)N_NODES * H1 * 4, stream);
  hipMemsetAsync(y1o, 0, (size_t)N_NODES * H1 * 4, stream);
  edge_prop_kernel<<<N_EDGES, 256, 0, stream>>>(t1i, ein,  win,  y1i, N_EDGES, H1);
  edge_prop_kernel<<<N_EDGES, 256, 0, stream>>>(t1o, eout, wout, y1o, N_EDGES, H1);
  layer_combine_kernel<<<N_NODES * H1 / 256, 256, 0, stream>>>(
      y1i, y1o, b1mi, b1si, b1mo, b1so, C1i, C1o, h1b, nullptr, H1);

  // --- layer 2 (embed written pre-tanh into d_out tail)
  float* out   = (float*)d_out;
  float* embed = out + (size_t)N_NODES * N_NODES;
  gemm(h1b, w2ci, nullptr, t2i, N_NODES, H2, H1);
  gemm(h1b, w2co, nullptr, t2o, N_NODES, H2, H1);
  hipMemsetAsync(y2i, 0, (size_t)N_NODES * H2 * 4, stream);
  hipMemsetAsync(y2o, 0, (size_t)N_NODES * H2 * 4, stream);
  edge_prop_kernel<<<N_EDGES, 128, 0, stream>>>(t2i, ein,  win,  y2i, N_EDGES, H2);
  edge_prop_kernel<<<N_EDGES, 128, 0, stream>>>(t2o, eout, wout, y2o, N_EDGES, H2);
  layer_combine_kernel<<<N_NODES * H2 / 256, 256, 0, stream>>>(
      y2i, y2o, b2mi, b2si, b2mo, b2so, C2i, C2o, h2b, embed, H2);

  // --- decoder GEMM (25.6 GF, bf16 WMMA) + in-place log_softmax
  gemm(h2b, wdb, bd, out, N_NODES, N_NODES, H2);
  log_softmax_kernel<<<N_NODES, 256, 0, stream>>>(out, N_NODES);
}
